// HeteroGraphAttentionEmbedding_52819507806816
// MI455X (gfx1250) — matrix operand order repaired and checked by reading, not compiled, not run on hardware
//
#include <hip/hip_runtime.h>
#include <hip/hip_bf16.h>

#define N_NODES 50000
#define E_EDGES 200000
#define D 256
#define DT 128

#define GEMM_ROWS 80          // rows per block; 50000/80=625, 200000/80=2500 exact
#define LDS_STRIDE 264        // 256 elements + 8 elements (16B) pad per row -> bank-friendly

typedef __attribute__((ext_vector_type(16))) __bf16 v16bf;
typedef __attribute__((ext_vector_type(8)))  __bf16 v8bf;
typedef __attribute__((ext_vector_type(8)))  float  v8f;
typedef __attribute__((ext_vector_type(4)))  unsigned u32x4;
typedef __attribute__((ext_vector_type(4)))  int      i32x4;
typedef __attribute__((ext_vector_type(8)))  int      i32x8;

static __device__ __forceinline__ __bf16 f2bf(float f) {
    union { float f; unsigned u; } a; a.f = f;
    unsigned r = a.u + 0x7FFFu + ((a.u >> 16) & 1u);   // round-to-nearest-even
    union { unsigned short s; __bf16 b; } o; o.s = (unsigned short)(r >> 16);
    return o.b;
}
static __device__ __forceinline__ float bf2f(__bf16 b) {
    union { unsigned short s; __bf16 b; } i; i.b = b;
    union { unsigned u; float f; } o; o.u = ((unsigned)i.s) << 16;
    return o.f;
}

// ---------------- conversion kernels ----------------
__global__ void cvt_bf16_kernel(const float* __restrict__ src, __bf16* __restrict__ dst, int n) {
    int i = blockIdx.x * 256 + threadIdx.x;
    if (i < n) dst[i] = f2bf(src[i]);
}

// W row-major [K=256][N=256] (f32)  ->  Wt row-major [N=256][K=256] (bf16)
__global__ void transpose_cvt_kernel(const float* __restrict__ W, __bf16* __restrict__ Wt) {
    int i = blockIdx.x * 256 + threadIdx.x;   // 65536 total
    int k = i >> 8, n = i & 255;
    Wt[n * D + k] = f2bf(W[k * D + n]);
}

// edge_attr[e, 0:128]  = cos(rel_t * W_t + b_t), edge_attr[e,128:256] = msg
__global__ void edge_attr_kernel(const float* __restrict__ t, const float* __restrict__ lu,
                                 const int* __restrict__ ei, const float* __restrict__ msg,
                                 const float* __restrict__ Wt_time, const float* __restrict__ b_t,
                                 __bf16* __restrict__ out) {
    int i = blockIdx.x * 256 + threadIdx.x;       // grid = E blocks exactly
    int e = i >> 8, c = i & 255;
    float val;
    if (c < DT) {
        float rel = t[e] - lu[ei[e]];             // src = ei[0*E + e]; MEAN=0, STD=1
        val = cosf(rel * Wt_time[c] + b_t[c]);
    } else {
        val = msg[(size_t)e * DT + (c - DT)];
    }
    out[(size_t)e * D + c] = f2bf(val);
}

// ---------------- WMMA GEMM: out[M,256] = A_bf16[M,256] @ W[256,256] (+bias) ----------------
// Block = 256 threads (8 waves). Block computes 80 rows x 256 cols.
// A tile (80x256 bf16, row-padded) staged into LDS once per block via the Tensor Data Mover.
// Wave w owns column tiles 2w/2w+1; inner loop: 8 K-steps x 5 row-tiles x 2 col-tiles WMMAs.
__global__ void __launch_bounds__(256) wmma_gemm_kernel(
        const __bf16* __restrict__ A, const __bf16* __restrict__ Wt,
        const float* __restrict__ bias, float* __restrict__ outF, __bf16* __restrict__ outB) {
    __shared__ __bf16 ldsA[GEMM_ROWS * LDS_STRIDE];   // 42240 bytes

    const int lane = threadIdx.x & 31;
    const int wave = threadIdx.x >> 5;
    const int rowBase = blockIdx.x * GEMM_ROWS;
    const bool lo = (lane < 16);
    const int mrow = lane & 15;
    const int c0 = wave * 32 + (lane & 15);     // column (tile 2w)
    const int c1 = c0 + 16;                     // column (tile 2w+1)

    // ---- TDM: DMA the 80x256 bf16 A tile into LDS (wave 0 issues, all waves consume) ----
    if (threadIdx.x < 32) {
        unsigned long long ga = (unsigned long long)(const void*)(A + (size_t)rowBase * D);
        unsigned lds_off = (unsigned)(unsigned long long)(void*)ldsA;   // low 32 bits = LDS byte offset
        u32x4 g0;
        g0[0] = 1u;                                            // count=1, user mode, no gather
        g0[1] = lds_off;                                       // lds_addr
        g0[2] = (unsigned)ga;                                  // global_addr[31:0]
        g0[3] = ((unsigned)(ga >> 32) & 0x01FFFFFFu) | (2u << 30);  // global_addr[56:32] | type=2
        i32x8 g1;
        // data_size=2B (code1), pad_enable, pad_interval=512B (code6), pad_amount=4 dwords (code3)
        g1[0] = (int)((1u << 16) | (1u << 20) | (6u << 22) | (3u << 25));
        g1[1] = (int)((256u & 0xFFFFu) << 16);                 // tensor_dim0 = 256 (lo16)
        g1[2] = (int)(((256u >> 16) & 0xFFFFu) | ((80u & 0xFFFFu) << 16)); // dim0 hi | tensor_dim1=80 lo
        g1[3] = (int)(((80u >> 16) & 0xFFFFu) | (256u << 16)); // dim1 hi | tile_dim0 = 256
        g1[4] = (int)80u;                                      // tile_dim1 = 80, tile_dim2 = 0
        g1[5] = (int)256u;                                     // tensor_dim0_stride = 256 (lo32)
        g1[6] = 0;                                             // stride hi | dim1_stride lo
        g1[7] = 0;
        i32x4 gz = {0, 0, 0, 0};                               // groups 2/3: unused dims = 0
#if __clang_major__ >= 23
        i32x8 gz8 = {0, 0, 0, 0, 0, 0, 0, 0};
        __builtin_amdgcn_tensor_load_to_lds(g0, g1, gz, gz, gz8, 0);
#else
        __builtin_amdgcn_tensor_load_to_lds(g0, g1, gz, gz, 0);
#endif
        __builtin_amdgcn_s_wait_tensorcnt(0);
    }
    __syncthreads();

    v8f acc[5][2];
#pragma unroll
    for (int rt = 0; rt < 5; ++rt) { acc[rt][0] = (v8f){}; acc[rt][1] = (v8f){}; }

    const __bf16* b0row = Wt + (size_t)c0 * D;
    const __bf16* b1row = Wt + (size_t)c1 * D;

    for (int kk = 0; kk < 8; ++kk) {
        // B fragment: 32x16, lanes 0-15: K 0-15; lanes 16-31: K 16-31 (contiguous in Wt)
        int kb = kk * 32 + (lo ? 0 : 16);
        v8bf t0 = *(const v8bf*)(b0row + kb);
        v8bf t1 = *(const v8bf*)(b0row + kb + 8);
        v8bf t2 = *(const v8bf*)(b1row + kb);
        v8bf t3 = *(const v8bf*)(b1row + kb + 8);
        v16bf bf0, bf1;
#pragma unroll
        for (int i = 0; i < 8; ++i) {
            bf0[i] = t0[i]; bf0[i + 8] = t1[i];
            bf1[i] = t2[i]; bf1[i + 8] = t3[i];
        }

        // A fragments from LDS: lanes 0-15: K 0-15; lanes 16-31: K 16-31
        int ka = kk * 32 + (lo ? 0 : 16);
#pragma unroll
        for (int rt = 0; rt < 5; ++rt) {
            const __bf16* lrow = ldsA + (rt * 16 + mrow) * LDS_STRIDE;
            v8bf a_lo = *(const v8bf*)(lrow + ka);
            v8bf a_hi = *(const v8bf*)(lrow + ka + 8);
            v16bf af;
#pragma unroll
            for (int i = 0; i < 8; ++i) { af[i] = a_lo[i]; af[i + 8] = a_hi[i]; }

            acc[rt][0] = __builtin_amdgcn_wmma_f32_16x16x32_bf16(false, af, false, bf0, (short)0, acc[rt][0], false, false);
            acc[rt][1] = __builtin_amdgcn_wmma_f32_16x16x32_bf16(false, af, false, bf1, (short)0, acc[rt][1], false, false);
        }
    }

    // D layout: VGPR r -> M = r (lanes 0-15) or 8+r (lanes 16-31); N = lane&15 within tile
    const int rOff = lo ? 0 : 8;
    const float bb0 = bias ? bias[c0] : 0.f;
    const float bb1 = bias ? bias[c1] : 0.f;
#pragma unroll
    for (int rt = 0; rt < 5; ++rt) {
#pragma unroll
        for (int r = 0; r < 8; ++r) {
            int row = rowBase + rt * 16 + r + rOff;
            float v0 = acc[rt][0][r] + bb0;
            float v1 = acc[rt][1][r] + bb1;
            if (outF) {
                outF[(size_t)row * D + c0] = v0;
                outF[(size_t)row * D + c1] = v1;
            } else {
                outB[(size_t)row * D + c0] = f2bf(v0);
                outB[(size_t)row * D + c1] = f2bf(v1);
            }
        }
    }
}

// ---------------- attention logits + segment max ----------------
// One wave per edge; lanes 0-15 = head 0 (cols 0..127), lanes 16-31 = head 1.
__global__ void __launch_bounds__(256) alpha_kernel(
        const int* __restrict__ ei, const float* __restrict__ q, const float* __restrict__ k,
        const __bf16* __restrict__ ebf, float* __restrict__ alpha, unsigned* __restrict__ segmax) {
    int lane = threadIdx.x & 31;
    int wv = threadIdx.x >> 5;
    int e = blockIdx.x * 8 + wv;                // grid = E/8 exactly
    int src = ei[e], dst = ei[E_EDGES + e];
    const float*  qr = q   + (size_t)dst * D;
    const float*  kr = k   + (size_t)src * D;
    const __bf16* er = ebf + (size_t)e * D;
    int cb = lane * 8;
    float s = 0.f;
#pragma unroll
    for (int i = 0; i < 8; ++i) {
        int c = cb + i;
        s += qr[c] * (kr[c] + bf2f(er[c]));
    }
    s += __shfl_xor(s, 8, 16);
    s += __shfl_xor(s, 4, 16);
    s += __shfl_xor(s, 2, 16);
    s += __shfl_xor(s, 1, 16);
    if ((lane & 15) == 0) {
        int h = lane >> 4;
        float a = s * 0.08838834764831845f;     // 1/sqrt(C), C=128
        alpha[e * 2 + h] = a;
        union { float f; unsigned u; } x; x.f = a;
        unsigned key = (x.u & 0x80000000u) ? ~x.u : (x.u | 0x80000000u);  // order-preserving
        atomicMax(&segmax[dst * 2 + h], key);
    }
}

__global__ void zero_kernel(unsigned* __restrict__ p, int n) {
    int i = blockIdx.x * 256 + threadIdx.x;
    if (i < n) p[i] = 0u;
}

// exp(alpha - max) and denominator accumulation
__global__ void expsum_kernel(const int* __restrict__ ei, float* __restrict__ alpha,
                              const unsigned* __restrict__ segmax, float* __restrict__ denom) {
    int i = blockIdx.x * 256 + threadIdx.x;
    if (i >= E_EDGES * 2) return;
    int e = i >> 1, h = i & 1;
    int dst = ei[E_EDGES + e];
    unsigned kkey = segmax[dst * 2 + h];
    unsigned ub = (kkey & 0x80000000u) ? (kkey & 0x7FFFFFFFu) : ~kkey;   // decode
    union { unsigned u; float f; } m; m.u = ub;
    float mx = __builtin_isfinite(m.f) ? m.f : 0.f;   // matches jnp.where(isfinite, amax, 0)
    float ex = expf(alpha[i] - mx);
    alpha[i] = ex;
    atomicAdd(&denom[dst * 2 + h], ex);
}

// out[dst] += (v[src] + e) * ex / (denom + 1e-16)
__global__ void msg_kernel(const int* __restrict__ ei, const float* __restrict__ v,
                           const __bf16* __restrict__ ebf, const float* __restrict__ alpha,
                           const float* __restrict__ denom, float* __restrict__ out) {
    int i = blockIdx.x * 256 + threadIdx.x;    // grid = E blocks exactly
    int e = i >> 8, c = i & 255;
    int h = c >> 7;
    int src = ei[e], dst = ei[E_EDGES + e];
    float coeff = alpha[e * 2 + h] / (denom[dst * 2 + h] + 1e-16f);
    float val = (v[(size_t)src * D + c] + bf2f(ebf[(size_t)e * D + c])) * coeff;
    atomicAdd(&out[(size_t)dst * D + c], val);
}

// ---------------- launcher ----------------
extern "C" void kernel_launch(void* const* d_in, const int* in_sizes, int n_in,
                              void* d_out, int out_size, void* d_ws, size_t ws_size,
                              hipStream_t stream) {
    (void)in_sizes; (void)n_in; (void)out_size; (void)ws_size;

    const float* x_user  = (const float*)d_in[0];
    const float* x_item  = (const float*)d_in[1];
    const float* lu_user = (const float*)d_in[2];
    const float* lu_item = (const float*)d_in[3];
    const int*   ei_a    = (const int*)d_in[4];
    const int*   ei_b    = (const int*)d_in[5];
    const float* t_a     = (const float*)d_in[6];
    const float* t_b     = (const float*)d_in[7];
    const float* msg_a   = (const float*)d_in[8];
    const float* msg_b   = (const float*)d_in[9];
    const float* W_t     = (const float*)d_in[10];
    const float* b_t     = (const float*)d_in[11];

    float* out_user = (float*)d_out;
    float* out_item = (float*)d_out + (size_t)N_NODES * D;

    // workspace carve-out
    char* p = (char*)d_ws;
    auto alloc = [&](size_t bytes) -> void* {
        void* r = (void*)p;
        p += (bytes + 255) & ~(size_t)255;
        return r;
    };
    __bf16* xb_user = (__bf16*)alloc((size_t)N_NODES * D * 2);
    __bf16* xb_item = (__bf16*)alloc((size_t)N_NODES * D * 2);
    __bf16* WT[5];
    for (int i = 0; i < 5; ++i) WT[i] = (__bf16*)alloc((size_t)D * D * 2);
    __bf16* eattr = (__bf16*)alloc((size_t)E_EDGES * D * 2);
    __bf16* eout  = (__bf16*)alloc((size_t)E_EDGES * D * 2);
    float* qbuf  = (float*)alloc((size_t)N_NODES * D * 4);
    float* kbuf  = (float*)alloc((size_t)N_NODES * D * 4);
    float* vbuf  = (float*)alloc((size_t)N_NODES * D * 4);
    float* alpha = (float*)alloc((size_t)E_EDGES * 2 * 4);
    unsigned* segmax = (unsigned*)alloc((size_t)N_NODES * 2 * 4);
    float* denom = (float*)alloc((size_t)N_NODES * 2 * 4);

    const int ND = N_NODES * D;
    cvt_bf16_kernel<<<(ND + 255) / 256, 256, 0, stream>>>(x_user, xb_user, ND);
    cvt_bf16_kernel<<<(ND + 255) / 256, 256, 0, stream>>>(x_item, xb_item, ND);

    // per-type descriptors: a = (user -> item), b = (item -> user)
    const int*   EI[2]  = { ei_a, ei_b };
    const float* TT[2]  = { t_a, t_b };
    const float* LU[2]  = { lu_user, lu_item };
    const float* MSG[2] = { msg_a, msg_b };
    const __bf16* XS[2] = { xb_user, xb_item };
    const __bf16* XD[2] = { xb_item, xb_user };
    float* OUT[2] = { out_item, out_user };
    const int WBASE[2] = { 12, 21 };   // Wq,bq,Wk,bk,Wv,bv,We,Ws,bs

    for (int ty = 0; ty < 2; ++ty) {
        const float* Wq = (const float*)d_in[WBASE[ty] + 0];
        const float* bq = (const float*)d_in[WBASE[ty] + 1];
        const float* Wk = (const float*)d_in[WBASE[ty] + 2];
        const float* bk = (const float*)d_in[WBASE[ty] + 3];
        const float* Wv = (const float*)d_in[WBASE[ty] + 4];
        const float* bv = (const float*)d_in[WBASE[ty] + 5];
        const float* We = (const float*)d_in[WBASE[ty] + 6];
        const float* Ws = (const float*)d_in[WBASE[ty] + 7];
        const float* bs = (const float*)d_in[WBASE[ty] + 8];

        transpose_cvt_kernel<<<256, 256, 0, stream>>>(Wq, WT[0]);
        transpose_cvt_kernel<<<256, 256, 0, stream>>>(Wk, WT[1]);
        transpose_cvt_kernel<<<256, 256, 0, stream>>>(Wv, WT[2]);
        transpose_cvt_kernel<<<256, 256, 0, stream>>>(We, WT[3]);
        transpose_cvt_kernel<<<256, 256, 0, stream>>>(Ws, WT[4]);

        edge_attr_kernel<<<E_EDGES, 256, 0, stream>>>(TT[ty], LU[ty], EI[ty], MSG[ty], W_t, b_t, eattr);

        // q (dst), k (src), v (src), edge projection, skip (-> output base)
        wmma_gemm_kernel<<<N_NODES / GEMM_ROWS, 256, 0, stream>>>(XD[ty], WT[0], bq, qbuf, nullptr);
        wmma_gemm_kernel<<<N_NODES / GEMM_ROWS, 256, 0, stream>>>(XS[ty], WT[1], bk, kbuf, nullptr);
        wmma_gemm_kernel<<<N_NODES / GEMM_ROWS, 256, 0, stream>>>(XS[ty], WT[2], bv, vbuf, nullptr);
        wmma_gemm_kernel<<<E_EDGES / GEMM_ROWS, 256, 0, stream>>>(eattr, WT[3], nullptr, nullptr, eout);
        wmma_gemm_kernel<<<N_NODES / GEMM_ROWS, 256, 0, stream>>>(XD[ty], WT[4], bs, OUT[ty], nullptr);

        zero_kernel<<<(2 * N_NODES + 255) / 256, 256, 0, stream>>>(segmax, 2 * N_NODES);
        zero_kernel<<<(2 * N_NODES + 255) / 256, 256, 0, stream>>>((unsigned*)denom, 2 * N_NODES);

        alpha_kernel<<<E_EDGES / 8, 256, 0, stream>>>(EI[ty], qbuf, kbuf, eout, alpha, segmax);
        expsum_kernel<<<(2 * E_EDGES + 255) / 256, 256, 0, stream>>>(EI[ty], alpha, segmax, denom);
        msg_kernel<<<E_EDGES, 256, 0, stream>>>(EI[ty], vbuf, eout, alpha, denom, OUT[ty]);
    }
}
